// Seq2Seq_4088808866017
// MI455X (gfx1250) — matrix operand order repaired and checked by reading, not compiled, not run on hardware
//
#include <hip/hip_runtime.h>
#include <hip/hip_bf16.h>

typedef __bf16 bf16_t;
typedef __attribute__((ext_vector_type(16))) __bf16 v16bf;
typedef __attribute__((ext_vector_type(8)))  __bf16 v8bf;
typedef __attribute__((ext_vector_type(8)))  float  v8f;

#define BB    16
#define SS    128
#define TT    128
#define EE    512
#define HH    1024
#define G4H   4096
#define VOUTN 32000
#define KCAT  (EE + HH + HH)   /* 2560: [x(512) | a(1024) | h(1024)] */

// ---------------------------------------------------------------------------
// Generic bf16 WMMA GEMM:  C[M,N] = A[M,K] * B[N,K]^T (+bias +addsrc, opt tanh)
// One wave32 per 16x16 output tile; 16x16x32 bf16 WMMA, fp32 accum.
// ---------------------------------------------------------------------------
__global__ __launch_bounds__(256) void wmma_gemm_bf16(
    const bf16_t* __restrict__ A, int lda,
    const bf16_t* __restrict__ Bw, int ldb,
    const float*  __restrict__ bias,
    const float*  __restrict__ addsrc, int ld_add,
    float*  __restrict__ C,   long long ldc,
    bf16_t* __restrict__ Cbf, int ldcbf,
    int Mtiles, int Ntiles, int K, int flags)
{
    int wave = (blockIdx.x * blockDim.x + threadIdx.x) >> 5;   // wave32
    int lane = threadIdx.x & 31;
    if (wave >= Mtiles * Ntiles) return;
    int mt = wave / Ntiles, nt = wave % Ntiles;
    int m0 = mt << 4, n0 = nt << 4;
    int g  = lane >> 4;        // half-wave (0/1)
    int nl = lane & 15;

    // A lane row = M=nl (both half-waves cover M=0..15, different K groups)
    const bf16_t* Arow = A  + (size_t)(m0 + nl) * lda;
    // B lane col = N=nl; half-wave selects K sub-block of 16
    const bf16_t* Brow = Bw + (size_t)(n0 + nl) * ldb + g * 16;

    v8f acc = {0.f, 0.f, 0.f, 0.f, 0.f, 0.f, 0.f, 0.f};

    union AU { v16bf v; v8bf h[2]; };

    for (int k0 = 0; k0 < K; k0 += 128) {
        __builtin_prefetch((const void*)(Brow + k0 + 512), 0, 1); // -> global_prefetch_b8
#pragma unroll
        for (int u = 0; u < 4; ++u) {
            int k = k0 + u * 32;
            // A k order used here (contiguous, matches B): lanes 0-15 K=[k..k+15], lanes 16-31 K=[k+16..k+31]
            AU au;
            au.h[0] = *(const v8bf*)(Arow + k + g * 16);
            au.h[1] = *(const v8bf*)(Arow + k + g * 16 + 8);
            // B k order used here (contiguous, matches A): lanes 0-15 K=[k..k+15], lanes 16-31 K=[k+16..k+31]
            v16bf bv = *(const v16bf*)(Brow + k);
            acc = __builtin_amdgcn_wmma_f32_16x16x32_bf16(
                      false, au.v, false, bv, (short)0, acc, false, false);
        }
    }

    // D layout: vgpr i, lane l -> M = i + 8*(l>=16), N = l%16
    int n = n0 + nl;
    float bval = bias ? bias[n] : 0.f;
#pragma unroll
    for (int i = 0; i < 8; ++i) {
        int r = m0 + i + 8 * g;
        float v = acc[i] + bval;
        if (addsrc) v += addsrc[(size_t)r * ld_add + n];
        if (flags & 1) v = tanhf(v);
        if (C)   C[(size_t)r * (size_t)ldc + n] = v;
        if (Cbf) Cbf[(size_t)r * ldcbf + n] = (bf16_t)v;
    }
}

// ---------------------------------------------------------------------------
// Helpers
// ---------------------------------------------------------------------------
__global__ void k_f2bf(const float* __restrict__ s, bf16_t* __restrict__ d, int n) {
    int i = blockIdx.x * blockDim.x + threadIdx.x;
    if (i < n) d[i] = (bf16_t)s[i];
}

__global__ void k_zero_f(float* p, int n) {
    int i = blockIdx.x * blockDim.x + threadIdx.x;
    if (i < n) p[i] = 0.f;
}
__global__ void k_zero_bf(bf16_t* p, int n) {
    int i = blockIdx.x * blockDim.x + threadIdx.x;
    if (i < n) p[i] = (bf16_t)0.f;
}
__global__ void k_addbias(const float* a, const float* b, float* o, int n) {
    int i = blockIdx.x * blockDim.x + threadIdx.x;
    if (i < n) o[i] = a[i] + b[i];
}

// dec_Wcat[n, 0:1536] = dec_Wih[n]; dec_Wcat[n, 1536:2560] = dec_Whh[n]
__global__ void k_cat_dec_w(const float* __restrict__ Wih, const float* __restrict__ Whh,
                            bf16_t* __restrict__ dst) {
    int i = blockIdx.x * blockDim.x + threadIdx.x;
    if (i >= G4H * KCAT) return;
    int n = i / KCAT, j = i % KCAT;
    float v = (j < (EE + HH)) ? Wih[(size_t)n * (EE + HH) + j]
                              : Whh[(size_t)n * HH + (j - (EE + HH))];
    dst[i] = (bf16_t)v;
}

// xe_bf time-major: row = t*B + b, so per-step slices of pregates are contiguous
__global__ void k_gather_xe(const int* __restrict__ toks, const float* __restrict__ emb,
                            bf16_t* __restrict__ xe) {
    int i = blockIdx.x * blockDim.x + threadIdx.x;
    if (i >= SS * BB * EE) return;
    int e = i % EE, row = i / EE;
    int t = row / BB, b = row % BB;
    int tok = toks[b * SS + t];
    xe[i] = (bf16_t)emb[(size_t)tok * EE + e];
}

// mask2 = mask with mask2[b, argmin(mask[b]) - 1 (wrap)] = 0
__global__ void k_mask2(const int* __restrict__ mask, int* __restrict__ m2) {
    int b = threadIdx.x;
    if (b >= BB) return;
    int len = 0, found = 0;
    for (int s = 0; s < SS; ++s) {
        int v = mask[b * SS + s];
        m2[b * SS + s] = v;
        if (!found && v == 0) { len = s; found = 1; }
    }
    int idx = len - 1; if (idx < 0) idx += SS;   // negative index wraps (JAX semantics)
    m2[b * SS + idx] = 0;
}

// LSTM cell elementwise; writes h (f32), c, bf16 h at given stride, opt enc_out slice
__global__ void k_lstm(const float* __restrict__ z, float* __restrict__ c,
                       float* __restrict__ h, bf16_t* __restrict__ hbf, int hbf_stride,
                       float* __restrict__ eout, long long eout_stride) {
    int i = blockIdx.x * blockDim.x + threadIdx.x;
    if (i >= BB * HH) return;
    int b = i / HH, j = i % HH;
    const float* zr = z + (size_t)b * G4H;
    float zi = zr[j], zf = zr[HH + j], zg = zr[2 * HH + j], zo = zr[3 * HH + j];
    float si = 1.f / (1.f + __expf(-zi));
    float sf = 1.f / (1.f + __expf(-zf));
    float so = 1.f / (1.f + __expf(-zo));
    float c2 = sf * c[i] + si * tanhf(zg);
    float h2 = so * tanhf(c2);
    c[i] = c2; h[i] = h2;
    hbf[(size_t)b * hbf_stride + j] = (bf16_t)h2;
    if (eout) eout[(size_t)b * eout_stride + j] = h2;
}

// decoder init: concatA h-slot = bf16(hT), a-slot = 0
__global__ void k_dec_init(const bf16_t* __restrict__ hbf, bf16_t* __restrict__ catA) {
    int i = blockIdx.x * blockDim.x + threadIdx.x;
    if (i >= BB * HH) return;
    int b = i / HH, j = i % HH;
    catA[(size_t)b * KCAT + EE + HH + j] = hbf[i];
    catA[(size_t)b * KCAT + EE + j]      = (bf16_t)0.f;
}

// decoder step embedding gather -> concatA x-slot
__global__ void k_dec_emb(const int* __restrict__ toks, const float* __restrict__ emb,
                          bf16_t* __restrict__ catA, int t) {
    int i = blockIdx.x * blockDim.x + threadIdx.x;
    if (i >= BB * EE) return;
    int b = i / EE, e = i % EE;
    int tok = toks[b * TT + t];
    catA[(size_t)b * KCAT + e] = (bf16_t)emb[(size_t)tok * EE + e];
}

// masked softmax attention; writes attnA = bf16([ct | h2]) per batch row
__global__ __launch_bounds__(128) void k_attn(const float* __restrict__ eout,
                                              const float* __restrict__ h2,
                                              const int* __restrict__ m2,
                                              bf16_t* __restrict__ attnA) {
    int b = blockIdx.x, s = threadIdx.x;
    __shared__ float sc[SS];
    __shared__ float red[SS];
    const float* eo = eout + ((size_t)b * SS + s) * HH;
    const float* hb = h2 + (size_t)b * HH;
    float acc = 0.f;
#pragma unroll 4
    for (int j = 0; j < HH; ++j) acc += eo[j] * hb[j];
    if (m2[b * SS + s] == 1) acc = -1e10f;
    sc[s] = acc; red[s] = acc;
    __syncthreads();
    for (int o = 64; o > 0; o >>= 1) { if (s < o) red[s] = fmaxf(red[s], red[s + o]); __syncthreads(); }
    float mx = red[0];
    __syncthreads();
    float ex = __expf(sc[s] - mx);
    red[s] = ex;
    __syncthreads();
    for (int o = 64; o > 0; o >>= 1) { if (s < o) red[s] += red[s + o]; __syncthreads(); }
    float inv = 1.f / red[0];
    __syncthreads();
    sc[s] = ex * inv;
    __syncthreads();
    for (int j = s; j < HH; j += SS) {
        float cv = 0.f;
#pragma unroll 4
        for (int q = 0; q < SS; ++q) cv += sc[q] * eout[((size_t)b * SS + q) * HH + j];
        attnA[(size_t)b * (2 * HH) + j]      = (bf16_t)cv;
        attnA[(size_t)b * (2 * HH) + HH + j] = (bf16_t)hb[j];
    }
}

// ---------------------------------------------------------------------------
extern "C" void kernel_launch(void* const* d_in, const int* in_sizes, int n_in,
                              void* d_out, int out_size, void* d_ws, size_t ws_size,
                              hipStream_t stream) {
    const int*   input_tokens  = (const int*)d_in[0];
    const int*   output_tokens = (const int*)d_in[1];
    const int*   mask          = (const int*)d_in[2];
    const float* emb_in  = (const float*)d_in[3];
    const float* emb_out = (const float*)d_in[4];
    const float* enc_Wih = (const float*)d_in[5];
    const float* enc_Whh = (const float*)d_in[6];
    const float* enc_bih = (const float*)d_in[7];
    const float* enc_bhh = (const float*)d_in[8];
    const float* dec_Wih = (const float*)d_in[9];
    const float* dec_Whh = (const float*)d_in[10];
    const float* dec_bih = (const float*)d_in[11];
    const float* dec_bhh = (const float*)d_in[12];
    const float* attn_W  = (const float*)d_in[13];
    const float* attn_b  = (const float*)d_in[14];
    const float* out_W   = (const float*)d_in[15];
    const float* out_b   = (const float*)d_in[16];
    float* out = (float*)d_out;
    (void)in_sizes; (void)n_in; (void)out_size; (void)ws_size;

    char* wptr = (char*)d_ws;
    auto carve = [&](size_t bytes) -> void* {
        void* p = (void*)wptr; wptr += (bytes + 255) & ~(size_t)255; return p;
    };
    bf16_t* enc_Wih_bf = (bf16_t*)carve((size_t)G4H * EE * 2);
    bf16_t* enc_Whh_bf = (bf16_t*)carve((size_t)G4H * HH * 2);
    bf16_t* dec_Wcat   = (bf16_t*)carve((size_t)G4H * KCAT * 2);
    bf16_t* attn_W_bf  = (bf16_t*)carve((size_t)HH * 2 * HH * 2);
    bf16_t* out_W_bf   = (bf16_t*)carve((size_t)VOUTN * HH * 2);
    bf16_t* xe_bf      = (bf16_t*)carve((size_t)SS * BB * EE * 2);
    float*  pregates   = (float*) carve((size_t)SS * BB * G4H * 4);
    float*  enc_out    = (float*) carve((size_t)BB * SS * HH * 4);
    float*  zbuf       = (float*) carve((size_t)BB * G4H * 4);
    float*  hbuf       = (float*) carve((size_t)BB * HH * 4);
    float*  cbuf       = (float*) carve((size_t)BB * HH * 4);
    bf16_t* hbuf_bf    = (bf16_t*)carve((size_t)BB * HH * 2);
    bf16_t* concatA    = (bf16_t*)carve((size_t)BB * KCAT * 2);
    bf16_t* attnA      = (bf16_t*)carve((size_t)BB * 2 * HH * 2);
    float*  enc_bias   = (float*) carve((size_t)G4H * 4);
    float*  dec_bias   = (float*) carve((size_t)G4H * 4);
    int*    mask2      = (int*)   carve((size_t)BB * SS * 4);

    auto ew = [&](int n) { return dim3((unsigned)((n + 255) / 256)); };
    auto gemm = [&](const bf16_t* A, int lda, const bf16_t* Bw, int ldb,
                    const float* bias, const float* addsrc, int ld_add,
                    float* C, long long ldc, bf16_t* Cbf, int ldcbf,
                    int Mtiles, int Ntiles, int K, int flags) {
        int blocks = (Mtiles * Ntiles + 7) / 8;   // 8 waves (tiles) per 256-thread block
        wmma_gemm_bf16<<<blocks, 256, 0, stream>>>(A, lda, Bw, ldb, bias, addsrc, ld_add,
                                                   C, ldc, Cbf, ldcbf, Mtiles, Ntiles, K, flags);
    };

    // ---- one-time (per call) weight conversion / packing -------------------
    k_f2bf<<<ew(G4H * EE), 256, 0, stream>>>(enc_Wih, enc_Wih_bf, G4H * EE);
    k_f2bf<<<ew(G4H * HH), 256, 0, stream>>>(enc_Whh, enc_Whh_bf, G4H * HH);
    k_f2bf<<<ew(HH * 2 * HH), 256, 0, stream>>>(attn_W, attn_W_bf, HH * 2 * HH);
    k_f2bf<<<ew(VOUTN * HH), 256, 0, stream>>>(out_W, out_W_bf, VOUTN * HH);
    k_cat_dec_w<<<ew(G4H * KCAT), 256, 0, stream>>>(dec_Wih, dec_Whh, dec_Wcat);
    k_addbias<<<ew(G4H), 256, 0, stream>>>(enc_bih, enc_bhh, enc_bias, G4H);
    k_addbias<<<ew(G4H), 256, 0, stream>>>(dec_bih, dec_bhh, dec_bias, G4H);
    k_gather_xe<<<ew(SS * BB * EE), 256, 0, stream>>>(input_tokens, emb_in, xe_bf);
    k_mask2<<<1, BB, 0, stream>>>(mask, mask2);
    k_zero_f<<<ew(BB * HH), 256, 0, stream>>>(hbuf, BB * HH);
    k_zero_f<<<ew(BB * HH), 256, 0, stream>>>(cbuf, BB * HH);
    k_zero_bf<<<ew(BB * HH), 256, 0, stream>>>(hbuf_bf, BB * HH);

    // ---- hoisted encoder input projection: pregates = xe @ Wih^T + bias ----
    gemm(xe_bf, EE, enc_Wih_bf, EE, enc_bias, nullptr, 0,
         pregates, G4H, nullptr, 0, (SS * BB) / 16, G4H / 16, EE, 0);

    // ---- encoder recurrence ------------------------------------------------
    for (int t = 0; t < SS; ++t) {
        gemm(hbuf_bf, HH, enc_Whh_bf, HH, nullptr,
             pregates + (size_t)t * BB * G4H, G4H,
             zbuf, G4H, nullptr, 0, 1, G4H / 16, HH, 0);
        k_lstm<<<ew(BB * HH), 256, 0, stream>>>(zbuf, cbuf, hbuf, hbuf_bf, HH,
                                                enc_out + (size_t)t * HH, (long long)SS * HH);
    }

    // ---- decoder init: concatA = [x | a=0 | h=hT], c carries over ----------
    k_dec_init<<<ew(BB * HH), 256, 0, stream>>>(hbuf_bf, concatA);

    // ---- decoder recurrence ------------------------------------------------
    for (int t = 0; t < TT - 1; ++t) {
        k_dec_emb<<<ew(BB * EE), 256, 0, stream>>>(output_tokens, emb_out, concatA, t);
        // z = [x|a|h] @ [Wih|Whh]^T + bias   (single fused K=2560 GEMM)
        gemm(concatA, KCAT, dec_Wcat, KCAT, dec_bias, nullptr, 0,
             zbuf, G4H, nullptr, 0, 1, G4H / 16, KCAT, 0);
        // LSTM cell; bf16 h2 goes straight into concatA h-slot for next step
        k_lstm<<<ew(BB * HH), 256, 0, stream>>>(zbuf, cbuf, hbuf,
                                                concatA + EE + HH, KCAT, nullptr, 0);
        // attention -> attnA = bf16([ct | h2])
        k_attn<<<BB, SS, 0, stream>>>(enc_out, hbuf, mask2, attnA);
        // a2 = tanh([ct|h2] @ attn_W^T + b); bf16 result into concatA a-slot
        gemm(attnA, 2 * HH, attn_W_bf, 2 * HH, attn_b, nullptr, 0,
             nullptr, 0, concatA + EE, KCAT, 1, HH / 16, 2 * HH, 1);
        // logits = a2 @ out_W^T + out_b  -> d_out[:, t, :]
        gemm(concatA + EE, KCAT, out_W_bf, HH, out_b, nullptr, 0,
             out + (size_t)t * VOUTN, (long long)(TT - 1) * VOUTN, nullptr, 0,
             1, VOUTN / 16, HH, 0);
    }
}
